// DynamicA8W8Linear_26637387170058
// MI455X (gfx1250) — 8-bit matrix operand order repaired and checked by reading, compiled, not run on hardware
//
#include <hip/hip_runtime.h>

typedef __attribute__((ext_vector_type(8))) int v8i;
typedef __attribute__((ext_vector_type(4))) int v4i;
typedef __attribute__((ext_vector_type(2))) int v2i;

__device__ __forceinline__ v8i load_frag_a(const signed char* p, int imm) {
  union { v8i v; v4i q[2]; } f;
  f.q[0] = *(const v4i*)(p + imm);
  f.q[1] = *(const v4i*)(p + imm + 32);
  return f.v;
}

__device__ __forceinline__ v8i load_frag_b(const signed char* p, int imm) {
  union { v8i v; v4i q[2]; } f;
  f.q[0] = *(const v4i*)(p + imm);
  f.q[1] = *(const v4i*)(p + imm + 32);
  return f.v;
}

__global__ __launch_bounds__(256) void dynq_kernel(
    const float* __restrict__ x, const float* __restrict__ smooth,
    signed char* __restrict__ xq, float* __restrict__ xscale) {
  constexpr int DIN = 4096;
  __shared__ float red[256];
  const int row = blockIdx.x;
  const int tid = threadIdx.x;
  const float* xr = x + (size_t)row * DIN + tid * 16;
  const float* sr = smooth + tid * 16;

  float p[16];
  float mx = 0.0f;
#pragma unroll
  for (int i = 0; i < 4; ++i) {
    float4 a = ((const float4*)xr)[i];
    float4 s = ((const float4*)sr)[i];
    p[4 * i + 0] = a.x * s.x;
    p[4 * i + 1] = a.y * s.y;
    p[4 * i + 2] = a.z * s.z;
    p[4 * i + 3] = a.w * s.w;
  }
#pragma unroll
  for (int i = 0; i < 16; ++i) mx = fmaxf(mx, __builtin_fabsf(p[i]));

  red[tid] = mx;
  __syncthreads();
  for (int off = 128; off > 0; off >>= 1) {
    if (tid < off) red[tid] = fmaxf(red[tid], red[tid + off]);
    __syncthreads();
  }
  const float m = fmaxf(red[0], 1e-20f);
  const float inv = 127.0f / m;
  if (tid == 0) xscale[row] = m * (1.0f / 127.0f);

  v4i packed;
#pragma unroll
  for (int i = 0; i < 4; ++i) {
    int b[4];
#pragma unroll
    for (int j = 0; j < 4; ++j) {
      int q = (int)__builtin_rintf(p[4 * i + j] * inv);
      q = q > 127 ? 127 : (q < -128 ? -128 : q);
      b[j] = q & 255;
    }
    packed[i] = b[0] | (b[1] << 8) | (b[2] << 16) | (b[3] << 24);
  }
  *(v4i*)(xq + (size_t)row * DIN + tid * 16) = packed;
}

__global__ __launch_bounds__(256) void a8w8_wmma_kernel(
    const signed char* __restrict__ xq, const signed char* __restrict__ w,
    const float* __restrict__ xscale, const float* __restrict__ wscale,
    const float* __restrict__ bias, float* __restrict__ out) {
  constexpr int DIN = 4096, DOUT = 4096;
  constexpr int ROWB = 16 * DIN;
  const int lane = threadIdx.x & 31;
  const int wv = threadIdx.x >> 5;
  const int m0 = blockIdx.x * 128 + (wv >> 1) * 32;
  const int n0 = blockIdx.y * 128 + (wv & 1) * 64;
  const int rsel = lane & 15;
  const int h8 = (lane & 16) ? 8 : 0;
  const int h16 = (lane & 16) ? 16 : 0;

  const signed char* abase = xq + (size_t)(m0 + rsel) * DIN + h16;
  const signed char* bbase = w + (size_t)(n0 + rsel) * DIN + h16;

  v8i c[2][4] = {};
  v8i a0[2], b0[4], a1[2], b1[4];

  auto load_chunk = [&](v8i(&a)[2], v8i(&b)[4], int k) {
#pragma unroll
    for (int mi = 0; mi < 2; ++mi) a[mi] = load_frag_a(abase + k, mi * ROWB);
#pragma unroll
    for (int ni = 0; ni < 4; ++ni) b[ni] = load_frag_b(bbase + k, ni * ROWB);
  };
  auto mma = [&](v8i(&a)[2], v8i(&b)[4]) {
#pragma unroll
    for (int mi = 0; mi < 2; ++mi)
#pragma unroll
      for (int ni = 0; ni < 4; ++ni)
        c[mi][ni] = __builtin_amdgcn_wmma_i32_16x16x64_iu8(
             true, a[mi],  true, b[ni], c[mi][ni],
             false,  false);
  };

  load_chunk(a0, b0, 0);

  int k = 0;
  for (; k < DIN - 128; k += 128) {
    load_chunk(a1, b1, k + 64);
    mma(a0, b0);
    load_chunk(a0, b0, k + 128);
    mma(a1, b1);
  }
  load_chunk(a1, b1, k + 64);
  mma(a0, b0);
  mma(a1, b1);

  const int hi = lane >> 4;
  const int ncol = lane & 15;
  float xs[2][8];
#pragma unroll
  for (int mi = 0; mi < 2; ++mi)
#pragma unroll
    for (int r = 0; r < 8; ++r)
      xs[mi][r] = xscale[m0 + mi * 16 + 8 * hi + r];

#pragma unroll
  for (int ni = 0; ni < 4; ++ni) {
    const int n = n0 + ni * 16 + ncol;
    const float wsc = wscale[n];
    const float bs = bias[n];
#pragma unroll
    for (int mi = 0; mi < 2; ++mi) {
      float* orow = out + (size_t)(m0 + mi * 16 + 8 * hi) * DOUT + n;
#pragma unroll
      for (int r = 0; r < 8; ++r)
        orow[(size_t)r * DOUT] = (float)c[mi][ni][r] * xs[mi][r] * wsc + bs;
    }
  }
}

extern "C" void kernel_launch(void* const* d_in, const int* in_sizes, int n_in,
                              void* d_out, int out_size, void* d_ws, size_t ws_size,
                              hipStream_t stream) {
  const float* x = (const float*)d_in[0];
  const signed char* weight = (const signed char*)d_in[1];
  const float* smooth = (const float*)d_in[2];
  const float* wscale = (const float*)d_in[3];
  const float* bias = (const float*)d_in[4];
  float* out = (float*)d_out;

  const int DIN = 4096;
  const int DOUT = 4096;
  const int NTOK = in_sizes[0] / DIN;

  signed char* xq = (signed char*)d_ws;
  float* xscale = (float*)((char*)d_ws + (size_t)NTOK * DIN);

  dynq_kernel<<<NTOK, 256, 0, stream>>>(x, smooth, xq, xscale);

  dim3 grid(NTOK / 128, DOUT / 128);
  a8w8_wmma_kernel<<<grid, 256, 0, stream>>>(xq, weight, xscale, wscale, bias, out);
}
